// Head_70600672412104
// MI455X (gfx1250) — matrix operand order repaired and checked by reading, not compiled, not run on hardware
//
#include <hip/hip_runtime.h>

// ---------------------------------------------------------------------------
// Causal single-head attention for MI455X (gfx1250, wave32, WMMA).
// B=8, T=2048, C=1024, H=64.
//   Pass 1: q/k = x @ W  -> fp16 [B*T, H];  v = x @ Wv -> fp16 TRANSPOSED
//           [B][H][T] so the attention V-fragments are contiguous in memory.
//   Pass 2: flash attention (online softmax), fp16 WMMA, fp32 accum/out.
//           No block barriers: only per-wave LDS for the P C->A relayout,
//           ordered by wave_barrier + in-order DS semantics.
// ---------------------------------------------------------------------------

typedef _Float16 h16v16 __attribute__((ext_vector_type(16)));
typedef _Float16 h16v8  __attribute__((ext_vector_type(8)));
typedef float    f32v8  __attribute__((ext_vector_type(8)));
typedef float    f32v4  __attribute__((ext_vector_type(4)));

#define WMMA_F16(a, b, c)                                                     \
  __builtin_amdgcn_wmma_f32_16x16x32_f16(false, (a), false, (b), (short)0,    \
                                         (c), false, false)

constexpr int kB = 8;
constexpr int kT = 2048;
constexpr int kC = 1024;
constexpr int kH = 64;

// ---------------------------------------------------------------------------
// Kernel 1: GEMM [M,1024] x [1024,64] -> fp16.
//   transposeOut == 0: out[row*64 + h]            ([B*T, H])
//   transposeOut == 1: out[(b*64 + h)*2048 + t]   ([B][H][T], V for attention)
//
// ISA fragment layouts (CDNA5 ISA §7.12.2, wave32):
//   A (16x32 f16): lane L -> row M = L%16; element e -> K = (e>>3)*16 + (L/16)*8 + (e&7)
//   B (32x16 f16): lane L -> col N = L%16; element e -> K as for A (this code: K = (L/16)*16 + e for A and B)
//   C (16x16 f32): lane L, reg r -> row M = r + (L/16)*8, col N = L%16
// ---------------------------------------------------------------------------
__global__ __launch_bounds__(128) void qkv_proj_kernel(
    const float* __restrict__ x,   // [M, C]
    const float* __restrict__ W,   // [C, H]
    _Float16* __restrict__ out,
    int transposeOut)
{
  __shared__ _Float16 Wf[8 * 4 * 32 * 16];  // 32 KB: [ki][n][lane][e]

  const int tid  = threadIdx.x;
  const int lane = tid & 31;
  const int g    = lane >> 4;   // half-wave group
  const int ln   = lane & 15;
  const int wv   = tid >> 5;
  const int r0   = blockIdx.x * 64 + wv * 16;

  f32v8 acc[4];
  for (int n = 0; n < 4; ++n)
    for (int r = 0; r < 8; ++r) acc[n][r] = 0.0f;

  const float* xrow = x + (size_t)(r0 + ln) * kC;

  for (int kb = 0; kb < 4; ++kb) {            // 4 chunks of 256 K-values
    __syncthreads();
    // Stage W[kb*256 .. +256)[0..64) into LDS in B-fragment order.
    for (int i = tid; i < 256 * 64; i += 128) {
      const int kl = i >> 6;                  // 0..255 (K within chunk)
      const int h  = i & 63;                  // coalesced over h
      const float w = W[(size_t)(kb * 256 + kl) * kH + h];
      const int ki = kl >> 5;                 // which 32-K slab
      const int kr = kl & 31;
      const int gg = kr >> 4;
      const int ee = kr & 15;
      const int n  = h >> 4;
      const int L  = gg * 16 + (h & 15);
      Wf[(((ki * 4 + n) * 32) + L) * 16 + ee] = (_Float16)w;
    }
    __syncthreads();

    for (int ki = 0; ki < 8; ++ki) {
      const int base = kb * 256 + ki * 32;
      // A fragment: one contiguous 16-float run per lane, fp32 -> fp16.
      f32v4 x0 = *(const f32v4*)(xrow + base + g * 16);
      f32v4 x1 = *(const f32v4*)(xrow + base + g * 16 + 4);
      f32v4 x2 = *(const f32v4*)(xrow + base + g * 16 + 8);
      f32v4 x3 = *(const f32v4*)(xrow + base + g * 16 + 12);
      h16v16 a;
      for (int j = 0; j < 4; ++j) {
        a[j]      = (_Float16)x0[j];
        a[4 + j]  = (_Float16)x1[j];
        a[8 + j]  = (_Float16)x2[j];
        a[12 + j] = (_Float16)x3[j];
      }
      for (int n = 0; n < 4; ++n) {
        h16v16 bf = *(const h16v16*)&Wf[(((ki * 4 + n) * 32) + lane) * 16];
        acc[n] = WMMA_F16(a, bf, acc[n]);
      }
    }
  }

  if (!transposeOut) {
    for (int n = 0; n < 4; ++n)
      for (int r = 0; r < 8; ++r)
        out[(size_t)(r0 + r + g * 8) * kH + n * 16 + ln] = (_Float16)acc[n][r];
  } else {
    // V: [B][H][T]. For fixed (lane,n), the 8 accumulator rows are 8
    // consecutive t values -> one 16B store each.
    const int rowBase = r0 + g * 8;        // first t of this lane's rows
    const int bb = rowBase >> 11;          // batch (T = 2048)
    const int tt = rowBase & (kT - 1);
    for (int n = 0; n < 4; ++n) {
      h16v8 pk;
      for (int r = 0; r < 8; ++r) pk[r] = (_Float16)acc[n][r];
      *(h16v8*)&out[((size_t)bb * kH + n * 16 + ln) * kT + tt] = pk;
    }
  }
}

// ---------------------------------------------------------------------------
// Kernel 2: flash attention. Grid (T/128, B), block = 256 (8 independent
// waves). Wave w owns query rows [bx*128 + w*16, +16); per-wave causal trip
// count (no block barriers). Q pre-scaled by H^-0.5 = 0.125 (exact in fp16).
// K fragments: contiguous 32B from k[B*T,H]. V fragments: contiguous 32B
// from transposed v[B][H][T]. P goes through per-wave LDS (C -> A layout),
// ordered by wave_barrier (DS ops of one wave complete in order).
// ---------------------------------------------------------------------------
__global__ __launch_bounds__(256) void flash_attn_kernel(
    const _Float16* __restrict__ qb,   // [B*T, H] fp16
    const _Float16* __restrict__ kb,   // [B*T, H] fp16
    const _Float16* __restrict__ vbT,  // [B][H][T] fp16 (transposed V)
    float* __restrict__ out)           // [B*T, H] fp32
{
  __shared__ _Float16 Pb[8][16 * 32];  // per-wave P staging

  const int tid  = threadIdx.x;
  const int lane = tid & 31;
  const int g    = lane >> 4;
  const int ln   = lane & 15;
  const int wv   = tid >> 5;
  const int b    = blockIdx.y;
  const size_t bT = (size_t)b * kT;
  const int q0   = blockIdx.x * 128 + wv * 16;  // local query base (in batch)

  // Load Q fragments once (K-dim = H = 64 -> two 32-K slabs), scale by 1/8.
  h16v16 aq[2];
  {
    const _Float16* qrow = qb + (bT + q0 + ln) * kH;
    for (int ki = 0; ki < 2; ++ki) {
      h16v8 lo = *(const h16v8*)(qrow + ki * 32 + g * 16);
      h16v8 hi = *(const h16v8*)(qrow + ki * 32 + g * 16 + 8);
      for (int j = 0; j < 8; ++j) {
        aq[ki][j]     = lo[j] * (_Float16)0.125f;
        aq[ki][8 + j] = hi[j] * (_Float16)0.125f;
      }
    }
  }

  f32v8 acc[4];
  for (int n = 0; n < 4; ++n)
    for (int r = 0; r < 8; ++r) acc[n][r] = 0.0f;
  float m[8], l[8];
  for (int r = 0; r < 8; ++r) { m[r] = -__builtin_inff(); l[r] = 0.0f; }

  _Float16* myP = &Pb[wv][0];
  const _Float16* vbase = vbT + (size_t)b * kH * kT;
  const int ntile = (q0 + 47) >> 5;    // keys 0 .. q0+15 covered

  for (int t = 0; t < ntile; ++t) {
    const int s0 = t * 32;

    // S = Q K^T : two 16x16 C-fragments (keys s0..+15 and s0+16..+31).
    f32v8 S0, S1;
    for (int r = 0; r < 8; ++r) { S0[r] = 0.0f; S1[r] = 0.0f; }
    for (int ki = 0; ki < 2; ++ki) {
      h16v16 k0 = *(const h16v16*)(kb + (bT + s0 + ln) * kH + ki * 32 + g * 16);
      h16v16 k1 = *(const h16v16*)(kb + (bT + s0 + 16 + ln) * kH + ki * 32 + g * 16);
      S0 = WMMA_F16(aq[ki], k0, S0);
      S1 = WMMA_F16(aq[ki], k1, S1);
    }

    // Causal mask + online softmax (row = r + g*8 in C layout; 16-lane
    // row reductions via shfl_xor masks 1..8 stay within the half-wave).
    __builtin_amdgcn_wave_barrier();   // prior-iteration P reads done
    for (int r = 0; r < 8; ++r) {
      const int qi = q0 + r + g * 8;
      float s0v = (s0 + ln      <= qi) ? S0[r] : -__builtin_inff();
      float s1v = (s0 + 16 + ln <= qi) ? S1[r] : -__builtin_inff();
      float mt = fmaxf(s0v, s1v);
      for (int off = 1; off < 16; off <<= 1) mt = fmaxf(mt, __shfl_xor(mt, off));
      const float mn    = fmaxf(m[r], mt);
      const float alpha = __expf(m[r] - mn);
      const float p0    = __expf(s0v - mn);
      const float p1    = __expf(s1v - mn);
      float rs = p0 + p1;
      for (int off = 1; off < 16; off <<= 1) rs += __shfl_xor(rs, off);
      l[r] = l[r] * alpha + rs;
      m[r] = mn;
      for (int n = 0; n < 4; ++n) acc[n][r] *= alpha;
      myP[(r + g * 8) * 32 + ln]      = (_Float16)p0;
      myP[(r + g * 8) * 32 + 16 + ln] = (_Float16)p1;
    }
    __builtin_amdgcn_wave_barrier();   // P writes ordered before reads

    // O += P(16x32) @ V(32x64). P A-fragment via LDS; V frags direct from
    // the transposed global V (contiguous 32B per lane, L2-resident tile).
    h16v16 pf;
    {
      h16v8 lo = *(const h16v8*)&myP[ln * 32 + g * 16];
      h16v8 hi = *(const h16v8*)&myP[ln * 32 + g * 16 + 8];
      for (int j = 0; j < 8; ++j) { pf[j] = lo[j]; pf[8 + j] = hi[j]; }
    }
    for (int n = 0; n < 4; ++n) {
      h16v16 vf = *(const h16v16*)(vbase + (size_t)(n * 16 + ln) * kT + s0 + g * 16);
      acc[n] = WMMA_F16(pf, vf, acc[n]);
    }
  }

  // Epilogue: O /= l, store fp32.
  for (int r = 0; r < 8; ++r) {
    const float inv = 1.0f / l[r];
    const int row = q0 + r + g * 8;
    for (int n = 0; n < 4; ++n)
      out[(bT + row) * kH + n * 16 + ln] = acc[n][r] * inv;
  }
}

// ---------------------------------------------------------------------------
extern "C" void kernel_launch(void* const* d_in, const int* in_sizes, int n_in,
                              void* d_out, int out_size, void* d_ws,
                              size_t ws_size, hipStream_t stream) {
  (void)in_sizes; (void)n_in; (void)out_size;

  const float* x  = (const float*)d_in[0];
  const float* Wk = (const float*)d_in[1];
  const float* Wq = (const float*)d_in[2];
  const float* Wv = (const float*)d_in[3];
  float* out = (float*)d_out;

  const size_t M = (size_t)kB * kT;             // 16384 rows
  const size_t bufElems = M * kH;               // 1M fp16 per buffer
  if (ws_size < 3 * bufElems * sizeof(_Float16)) return;  // need 6 MB scratch

  _Float16* qbuf = (_Float16*)d_ws;
  _Float16* kbuf = qbuf + bufElems;
  _Float16* vbufT = kbuf + bufElems;            // [B][H][T]

  const dim3 blk(128);
  qkv_proj_kernel<<<dim3((unsigned)(M / 64)), blk, 0, stream>>>(x, Wq, qbuf, 0);
  qkv_proj_kernel<<<dim3((unsigned)(M / 64)), blk, 0, stream>>>(x, Wk, kbuf, 0);
  qkv_proj_kernel<<<dim3((unsigned)(M / 64)), blk, 0, stream>>>(x, Wv, vbufT, 1);
  flash_attn_kernel<<<dim3(kT / 128, kB), dim3(256), 0, stream>>>(qbuf, kbuf,
                                                                  vbufT, out);
}
